// PairwiseMLPLinkPredictor_68143951118895
// MI455X (gfx1250) — matrix operand order repaired and checked by reading, compiled, not run on hardware
//
#include <hip/hip_runtime.h>
#include <hip/hip_bf16.h>

typedef _Float16 v8h  __attribute__((ext_vector_type(8)));
typedef _Float16 v16h __attribute__((ext_vector_type(16)));
typedef float    v8f  __attribute__((ext_vector_type(8)));

#define N_NODES 8192
#define IN_CH   128
#define HIDDEN  256
#define H2      128
#define WORDS   256              // 8192 bits / 32 per adjacency row
#define K1      644              // 5*128 + 4
#define K1P     672              // padded to 21*32 for K-steps of 32
#define PPB     64               // pairs per block = 4 M-tiles
#define MT      (PPB / 16)       // 4 M-tiles
#define EPSV    1e-8f

// ---- workspace layout (bytes) ----
#define OFF_ADJ   0u
#define SZ_ADJ    (N_NODES * WORDS * 4u)          // 8 MB bitset adjacency
#define OFF_ILOG  (OFF_ADJ + SZ_ADJ)
#define OFF_IDEG  (OFF_ILOG + N_NODES * 4u)
#define OFF_W1T   (OFF_IDEG + N_NODES * 4u)
#define SZ_W1T    (HIDDEN * K1P * 2u)             // f16, N-major, K-padded
#define OFF_W2T   (OFF_W1T + SZ_W1T)
#define SZ_W2T    (H2 * HIDDEN * 2u)              // f16, N-major

// ---------------- phase A: adjacency + degree stats ----------------
__global__ void k_zero(unsigned* p, int n) {
  int i = blockIdx.x * blockDim.x + threadIdx.x;
  int stride = gridDim.x * blockDim.x;
  for (; i < n; i += stride) p[i] = 0u;
}

__global__ void k_scatter(const long long* ei, int E, unsigned* adj) {
  int e = blockIdx.x * blockDim.x + threadIdx.x;
  if (e >= E) return;
  int u = (int)ei[e];          // edge_index row 0
  int v = (int)ei[E + e];      // edge_index row 1
  if (u == v) return;          // diagonal zeroed in reference
  atomicOr(&adj[u * WORDS + (v >> 5)], 1u << (v & 31));
}

__global__ void k_degree(const unsigned* adj, float* ilog, float* ideg) {
  int n = blockIdx.x * blockDim.x + threadIdx.x;
  if (n >= N_NODES) return;
  const unsigned* row = adj + (size_t)n * WORDS;
  int d = 0;
  for (int j = 0; j < WORDS; ++j) d += __popc(row[j]);
  float fd = (float)d;
  ilog[n] = (d > 1) ? 1.0f / logf(fd) : 0.0f;
  ideg[n] = (d > 0) ? 1.0f / fd : 0.0f;
}

// ---------------- phase B: weight transpose/convert to f16 ----------------
__global__ void k_w1t(const float* W1, _Float16* W1t) {
  int i = blockIdx.x * blockDim.x + threadIdx.x;      // HIDDEN*K1P elems
  if (i >= HIDDEN * K1P) return;
  int n = i / K1P, k = i % K1P;
  W1t[(size_t)n * K1P + k] = (k < K1) ? (_Float16)W1[(size_t)k * HIDDEN + n]
                                      : (_Float16)0.0f;
}

__global__ void k_w2t(const float* W2, _Float16* W2t) {
  int i = blockIdx.x * blockDim.x + threadIdx.x;      // H2*HIDDEN elems
  if (i >= H2 * HIDDEN) return;
  int n = i / HIDDEN, k = i % HIDDEN;
  W2t[(size_t)n * HIDDEN + k] = (_Float16)W2[(size_t)k * H2 + n];
}

// ---------------- phase C: pair scoring ----------------
__device__ __forceinline__ v16h frag16(const _Float16* lo, const _Float16* hi) {
  v8h a = *(const v8h*)lo;
  v8h b = *(const v8h*)hi;
  return __builtin_shufflevector(a, b, 0,1,2,3,4,5,6,7,8,9,10,11,12,13,14,15);
}

#define WMMA16(A, B, C) \
  __builtin_amdgcn_wmma_f32_16x16x32_f16(false, (A), false, (B), (short)0, (C), false, false)

__launch_bounds__(256)
__global__ void k_score(const float* __restrict__ x,
                        const long long* __restrict__ pairs,
                        const unsigned* __restrict__ adj,
                        const float* __restrict__ ilog,
                        const float* __restrict__ ideg,
                        const _Float16* __restrict__ W1t, const float* __restrict__ b1,
                        const _Float16* __restrict__ W2t, const float* __restrict__ b2,
                        const float* __restrict__ W3, const float* __restrict__ b3,
                        float* __restrict__ out) {
  __shared__ int   sU[PPB], sV[PPB];
  __shared__ float sScal[PPB][4];                      // cos, cn, aa, ra
  // sFeat (86KB) and sH1 (32KB) live together during GEMM1;
  // sH2 (32KB f32) aliases the dead sFeat region after GEMM1.
  __shared__ __align__(16) char sBuf[PPB * K1P * 2 + PPB * HIDDEN * 2];
  _Float16* sFeat = (_Float16*)sBuf;
  _Float16* sH1   = (_Float16*)(sBuf + PPB * K1P * 2);
  float*    sH2   = (float*)sBuf;

  const int tid  = threadIdx.x;
  const int lane = tid & 31;
  const int wave = tid >> 5;       // 0..7
  const int g    = tid >> 4;       // group 0..15 (16 lanes each)
  const int gl   = tid & 15;       // lane within group
  const int p0   = blockIdx.x * PPB;

  if (tid < PPB) {
    sU[tid] = (int)pairs[2LL * (p0 + tid)];
    sV[tid] = (int)pairs[2LL * (p0 + tid) + 1];
  }
  __syncthreads();

  // ---- pair scalars (cos, cn, aa, ra): each 16-lane group does 4 pairs ----
  for (int pp = 0; pp < MT; ++pp) {
    const int i = pp * 16 + g;
    const int u = sU[i], v = sV[i];
    const float* xu = x + (size_t)u * IN_CH;
    const float* xv = x + (size_t)v * IN_CH;
    float dot = 0.f, nu2 = 0.f, nv2 = 0.f;
    #pragma unroll
    for (int q = 0; q < 8; ++q) {
      float a = xu[gl * 8 + q], b = xv[gl * 8 + q];
      dot += a * b; nu2 += a * a; nv2 += b * b;
    }
    const unsigned* Au = adj + (size_t)u * WORDS;
    const unsigned* Av = adj + (size_t)v * WORDS;
    float cn = 0.f, aa = 0.f, ra = 0.f;
    for (int j = gl; j < WORDS; j += 16) {
      unsigned w = Au[j] & Av[j];
      cn += (float)__popc(w);
      while (w) {                     // avg degree ~32 -> nearly empty
        int bit = __ffs(w) - 1;
        w &= (w - 1);
        int col = (j << 5) + bit;
        aa += ilog[col];
        ra += ideg[col];
      }
    }
    #pragma unroll
    for (int m = 8; m >= 1; m >>= 1) {
      dot += __shfl_xor(dot, m, 16);
      nu2 += __shfl_xor(nu2, m, 16);
      nv2 += __shfl_xor(nv2, m, 16);
      cn  += __shfl_xor(cn,  m, 16);
      aa  += __shfl_xor(aa,  m, 16);
      ra  += __shfl_xor(ra,  m, 16);
    }
    if (gl == 0) {
      float denom = fmaxf(sqrtf(nu2) * sqrtf(nv2), EPSV);
      sScal[i][0] = dot / denom;
      sScal[i][1] = cn;
      sScal[i][2] = aa;
      sScal[i][3] = ra;
    }
  }
  __syncthreads();

  // ---- build feats rows (f16): [xu | xv | xu*xv | |xu-xv| | xu+xv | 4 scalars | pad] ----
  for (int pp = 0; pp < MT; ++pp) {
    const int i = pp * 16 + g;
    const int u = sU[i], v = sV[i];
    const float* xu = x + (size_t)u * IN_CH;
    const float* xv = x + (size_t)v * IN_CH;
    _Float16* f = sFeat + (size_t)i * K1P;
    #pragma unroll
    for (int q = 0; q < 8; ++q) {
      int c = gl * 8 + q;
      float a = xu[c], b = xv[c];
      f[c]             = (_Float16)a;
      f[IN_CH + c]     = (_Float16)b;
      f[2 * IN_CH + c] = (_Float16)(a * b);
      f[3 * IN_CH + c] = (_Float16)fabsf(a - b);
      f[4 * IN_CH + c] = (_Float16)(a + b);
    }
    if (gl == 0) {
      f[640] = (_Float16)sScal[i][0];
      f[641] = (_Float16)sScal[i][1];
      f[642] = (_Float16)sScal[i][2];
      f[643] = (_Float16)sScal[i][3];
      for (int k = K1; k < K1P; ++k) f[k] = (_Float16)0.0f;   // K padding
    }
  }
  __syncthreads();

  const int half = lane >> 4;     // which 16-lane half of the wave
  const int nl   = lane & 15;

  // ---- GEMM1: [64 x 672]f16 @ [672 x 256]f16 -> relu -> sH1 (f16) ----
  // Register blocking: 4 M-tiles x 2 N-tiles per wave; each B frag feeds 4 WMMA.
  {
    const int n0 = wave * 32 + nl;        // N-tile 2*wave
    const int n1 = n0 + 16;               // N-tile 2*wave+1
    const float bias0 = b1[n0], bias1 = b1[n1];
    v8f acc0[MT], acc1[MT];
    #pragma unroll
    for (int mt = 0; mt < MT; ++mt)
      #pragma unroll
      for (int j = 0; j < 8; ++j) { acc0[mt][j] = bias0; acc1[mt][j] = bias1; }

    const _Float16* brow0 = W1t + (size_t)n0 * K1P + 16 * half;
    const _Float16* brow1 = W1t + (size_t)n1 * K1P + 16 * half;
    for (int kk = 0; kk < K1P / 32; ++kk) {
      const int ko = kk * 32;
      __builtin_prefetch(brow0 + ko + 64, 0, 1);
      // B frag: lanes 0-15 K=0..15, lanes 16-31 K=16..31 (N-major rows)
      v16h bb0 = frag16(brow0 + ko, brow0 + ko + 8);
      v16h bb1 = frag16(brow1 + ko, brow1 + ko + 8);
      #pragma unroll
      for (int mt = 0; mt < MT; ++mt) {
        // A frag: lanes 0-15 K=0..15, lanes 16-31 K=16..31
        const _Float16* arow = sFeat + (size_t)(mt * 16 + nl) * K1P;
        v16h a = frag16(arow + ko + 16 * half, arow + ko + 16 * half + 8);
        acc0[mt] = WMMA16(a, bb0, acc0[mt]);
        acc1[mt] = WMMA16(a, bb1, acc1[mt]);
      }
    }
    #pragma unroll
    for (int mt = 0; mt < MT; ++mt)
      #pragma unroll
      for (int j = 0; j < 8; ++j) {
        int m = mt * 16 + j + 8 * half;   // C/D layout: M = vgpr + 8*(lane/16)
        sH1[(size_t)m * HIDDEN + n0] = (_Float16)fmaxf(acc0[mt][j], 0.0f);
        sH1[(size_t)m * HIDDEN + n1] = (_Float16)fmaxf(acc1[mt][j], 0.0f);
      }
  }
  __syncthreads();

  // ---- GEMM2: [64 x 256]f16 @ [256 x 128]f16 -> relu -> sH2 (f32) ----
  {
    const int n = wave * 16 + nl;         // 8 waves cover 128 outputs
    const float bias = b2[n];
    v8f acc[MT];
    #pragma unroll
    for (int mt = 0; mt < MT; ++mt)
      #pragma unroll
      for (int j = 0; j < 8; ++j) acc[mt][j] = bias;

    const _Float16* brow = W2t + (size_t)n * HIDDEN + 16 * half;
    for (int kk = 0; kk < HIDDEN / 32; ++kk) {
      const int ko = kk * 32;
      v16h bb = frag16(brow + ko, brow + ko + 8);
      #pragma unroll
      for (int mt = 0; mt < MT; ++mt) {
        const _Float16* arow = sH1 + (size_t)(mt * 16 + nl) * HIDDEN;
        v16h a = frag16(arow + ko + 16 * half, arow + ko + 16 * half + 8);
        acc[mt] = WMMA16(a, bb, acc[mt]);
      }
    }
    #pragma unroll
    for (int mt = 0; mt < MT; ++mt)
      #pragma unroll
      for (int j = 0; j < 8; ++j) {
        int m = mt * 16 + j + 8 * half;
        sH2[(size_t)m * H2 + n] = fmaxf(acc[mt][j], 0.0f);
      }
  }
  __syncthreads();

  // ---- GEMM3: [64 x 128] @ [128 x 1] + b3 ----
  for (int pp = 0; pp < MT; ++pp) {
    const int i = pp * 16 + g;
    const float* h = sH2 + (size_t)i * H2;
    float s = 0.f;
    #pragma unroll
    for (int q = 0; q < 8; ++q) s += h[gl * 8 + q] * W3[gl * 8 + q];
    #pragma unroll
    for (int m = 8; m >= 1; m >>= 1) s += __shfl_xor(s, m, 16);
    if (gl == 0) out[p0 + i] = s + b3[0];
  }
}

// ---------------- host launch ----------------
extern "C" void kernel_launch(void* const* d_in, const int* in_sizes, int n_in,
                              void* d_out, int out_size, void* d_ws, size_t ws_size,
                              hipStream_t stream) {
  const float*     x  = (const float*)d_in[0];
  const long long* ei = (const long long*)d_in[1];   // int64 (2, E)
  const long long* ep = (const long long*)d_in[2];   // int64 (P, 2)
  const float*     W1 = (const float*)d_in[3];
  const float*     b1 = (const float*)d_in[4];
  const float*     W2 = (const float*)d_in[5];
  const float*     b2 = (const float*)d_in[6];
  const float*     W3 = (const float*)d_in[7];
  const float*     b3 = (const float*)d_in[8];
  float*           out = (float*)d_out;

  char* ws = (char*)d_ws;
  unsigned*  adj  = (unsigned*)(ws + OFF_ADJ);
  float*     ilog = (float*)(ws + OFF_ILOG);
  float*     ideg = (float*)(ws + OFF_IDEG);
  _Float16*  W1t  = (_Float16*)(ws + OFF_W1T);
  _Float16*  W2t  = (_Float16*)(ws + OFF_W2T);

  const int E = in_sizes[1] / 2;
  const int P = in_sizes[2] / 2;

  k_zero   <<<2048, 256, 0, stream>>>(adj, N_NODES * WORDS);
  k_scatter<<<(E + 255) / 256, 256, 0, stream>>>(ei, E, adj);
  k_degree <<<(N_NODES + 255) / 256, 256, 0, stream>>>(adj, ilog, ideg);
  k_w1t    <<<(HIDDEN * K1P + 255) / 256, 256, 0, stream>>>(W1, W1t);
  k_w2t    <<<(H2 * HIDDEN + 255) / 256, 256, 0, stream>>>(W2, W2t);
  k_score  <<<P / PPB, 256, 0, stream>>>(x, ep, adj, ilog, ideg,
                                         W1t, b1, W2t, b2, W3, b3, out);
}
